// Attention_12463995093086
// MI455X (gfx1250) — matrix operand order repaired and checked by reading, not compiled, not run on hardware
//
#include <hip/hip_runtime.h>
#include <stdint.h>

// ---------------------------------------------------------------------------
// CDNA5 (gfx1250) bf16 WMMA attention pipeline, wave32.
// - v_wmma_f32_16x16x32_bf16 for all matmuls (f32 accumulate)
// - global_load_async_to_lds_b128 (ASYNCcnt) double-buffered K/V staging
// ---------------------------------------------------------------------------

typedef __attribute__((ext_vector_type(16))) __bf16 v16bf;
typedef __attribute__((ext_vector_type(8)))  float  v8f;

union FragBf {
    v16bf    h;
    uint32_t u[8];
};

__device__ inline uint16_t f2bfu(float f) {
    uint32_t u = __float_as_uint(f);
    u += 0x7FFFu + ((u >> 16) & 1u);       // round-to-nearest-even
    return (uint16_t)(u >> 16);
}

__device__ inline v8f v8f_zero() {
    v8f z = {0.f, 0.f, 0.f, 0.f, 0.f, 0.f, 0.f, 0.f};
    return z;
}

__device__ inline v8f wmma_bf16(const FragBf& a, const FragBf& b, v8f c) {
    return __builtin_amdgcn_wmma_f32_16x16x32_bf16(
        false, a.h, false, b.h, (short)0, c, false, false);
}

// Async 16-byte copy global -> LDS (ASYNCcnt). Low 32 bits of a generic LDS
// pointer are the wave-relative LDS byte offset (aperture-based flat addr).
__device__ inline void async_copy_b128(const void* gsrc, void* lds_dst) {
    uint32_t l = (uint32_t)(uintptr_t)lds_dst;
    uint64_t g = (uint64_t)(uintptr_t)gsrc;
    asm volatile("global_load_async_to_lds_b128 %0, %1, off"
                 :: "v"(l), "v"(g) : "memory");
}
__device__ inline void wait_async0() {
    asm volatile("s_wait_asynccnt 0x0" ::: "memory");
}

#define KC     768
#define HEADS  12
#define DH     64
#define NTOK   384
#define SEG_ELEMS 18874368ull  // 64*12*384*64 bf16 elems per Q/K/V buffer

// ---------------------------------------------------------------------------
// Kernel 1: qkv = concat(x_v, x_i) @ qkv_w^T  -> bf16 Q [b,h,n,d],
//           K^T [b,h,d,n], V [b,h,n,d]
// ---------------------------------------------------------------------------
__global__ __launch_bounds__(256) void qkv_gemm_kernel(
    const float* __restrict__ xv, const float* __restrict__ xi,
    const float* __restrict__ qkv_w,
    uint16_t* __restrict__ qws, uint16_t* __restrict__ kws,
    uint16_t* __restrict__ vws)
{
    __shared__ __align__(16) uint16_t As[128 * 40];
    __shared__ __align__(16) uint16_t Bs[32 * 136];

    const int t    = threadIdx.x;
    const int lane = t & 31, wave = t >> 5;
    const int half = lane >> 4, l16 = lane & 15;
    const int wm   = wave >> 1, wn = wave & 1;
    const int bm   = blockIdx.x, bn = blockIdx.y;

    // x = concat(x_v, x_i): split at row 12288 = block 96 -> uniform per block
    const float* xbase = (bm < 96)
        ? xv + (size_t)bm * 128 * KC
        : xi + ((size_t)bm * 128 - 12288) * KC;

    v8f acc[2][4];
#pragma unroll
    for (int mt = 0; mt < 2; ++mt)
#pragma unroll
        for (int nt = 0; nt < 4; ++nt) acc[mt][nt] = v8f_zero();

    const int ar = t >> 3;        // 0..31
    const int ac = (t & 7) * 4;   // 0..28 step 4

    for (int kt = 0; kt < 24; ++kt) {
        const int k0 = kt * 32;
#pragma unroll
        for (int rr = 0; rr < 4; ++rr) {
            int row = ar + rr * 32;
            const float* src = xbase + (size_t)row * KC + k0 + ac;
            float4 v = *(const float4*)src;
            uint32_t* d = (uint32_t*)&As[row * 40 + ac];
            d[0] = (uint32_t)f2bfu(v.x) | ((uint32_t)f2bfu(v.y) << 16);
            d[1] = (uint32_t)f2bfu(v.z) | ((uint32_t)f2bfu(v.w) << 16);
        }
#pragma unroll
        for (int nn = 0; nn < 4; ++nn) {
            int n  = ar + nn * 32;
            int gn = bn * 128 + n;
            const float* src = qkv_w + (size_t)gn * KC + k0 + ac;
            float4 v = *(const float4*)src;
            Bs[(ac + 0) * 136 + n] = f2bfu(v.x);
            Bs[(ac + 1) * 136 + n] = f2bfu(v.y);
            Bs[(ac + 2) * 136 + n] = f2bfu(v.z);
            Bs[(ac + 3) * 136 + n] = f2bfu(v.w);
        }
        if (kt + 1 < 24)   // gfx1250 global_prefetch_b8
            __builtin_prefetch(xbase + (size_t)ar * KC + k0 + 32 + ac, 0, 1);
        __syncthreads();

        FragBf a[2], b[4];
#pragma unroll
        for (int mt = 0; mt < 2; ++mt) {
            const uint16_t* ap = &As[(wm * 32 + mt * 16 + l16) * 40];
#pragma unroll
            for (int i = 0; i < 8; ++i) {
                int kk = ((i < 4) ? 0 : 16) + half * 8 + (i & 3) * 2;
                a[mt].u[i] = *(const uint32_t*)(ap + kk);
            }
        }
#pragma unroll
        for (int nt = 0; nt < 4; ++nt) {
            const uint16_t* bp = &Bs[wn * 64 + nt * 16 + l16];
#pragma unroll
            for (int i = 0; i < 8; ++i) {
                int kk = ((i < 4) ? 0 : 16) + half * 8 + (i & 3) * 2;
                b[nt].u[i] = (uint32_t)bp[kk * 136] | ((uint32_t)bp[(kk + 1) * 136] << 16);
            }
        }
#pragma unroll
        for (int mt = 0; mt < 2; ++mt)
#pragma unroll
            for (int nt = 0; nt < 4; ++nt)
                acc[mt][nt] = wmma_bf16(a[mt], b[nt], acc[mt][nt]);
        __syncthreads();
    }

    // ---- epilogue: block-uniform index math (SALU), scatter bf16 ----
    // cols: s = bn/6 (Q/K/V select), head hh = (bn%6)*2 + wn, d = nt*16+l16
    // rows: b2 = bm/3, n = (bm%3)*128 + wm*32 + mt*16 + v + 8*half
    const int    s     = bn / 6;
    const size_t hbase = (size_t)(bm / 3) * HEADS + (bn % 6) * 2 + wn;
    const int    nb    = (bm % 3) * 128 + wm * 32;

    if (s == 1) {  // K, transposed [b,h,d,n]
#pragma unroll
        for (int mt = 0; mt < 2; ++mt)
#pragma unroll
            for (int nt = 0; nt < 4; ++nt)
#pragma unroll
                for (int v = 0; v < 8; ++v) {
                    int n = nb + mt * 16 + v + 8 * half;
                    int d = nt * 16 + l16;
                    kws[(hbase * DH + d) * NTOK + n] = f2bfu(acc[mt][nt][v]);
                }
    } else {       // Q or V, [b,h,n,d]
        uint16_t* dst = (s == 0) ? qws : vws;
#pragma unroll
        for (int mt = 0; mt < 2; ++mt)
#pragma unroll
            for (int nt = 0; nt < 4; ++nt)
#pragma unroll
                for (int v = 0; v < 8; ++v) {
                    int n = nb + mt * 16 + v + 8 * half;
                    dst[(hbase * NTOK + n) * DH + nt * 16 + l16] = f2bfu(acc[mt][nt][v]);
                }
    }
}

// ---------------------------------------------------------------------------
// Kernel 2: flash attention with async double-buffered K/V staging in LDS.
// Block = (b2, head, 128-query block); 8 waves x 16 queries.
// ---------------------------------------------------------------------------
__global__ __launch_bounds__(256) void attn_kernel(
    const uint16_t* __restrict__ qws, const uint16_t* __restrict__ kws,
    const uint16_t* __restrict__ vws, uint16_t* __restrict__ aws)
{
    __shared__ __align__(16) uint16_t Kl[2][64 * 40];  // [d][key] pad 40
    __shared__ __align__(16) uint16_t Vl[2][32 * 72];  // [key][d] pad 72
    __shared__ __align__(16) uint16_t Pl[8 * 16 * 32]; // per-wave P staging

    const int t    = threadIdx.x;
    const int lane = t & 31, wave = t >> 5;
    const int half = lane >> 4, l16 = lane & 15;
    const int bid  = blockIdx.x;
    const int qb   = bid % 3;
    const int h    = (bid / 3) % HEADS;
    const int b2   = bid / (3 * HEADS);
    const int q0   = qb * 128 + wave * 16;
    const float scale = 0.125f;  // Dh^-0.5

    // Q fragments: 16 queries x 64 dims = two k=32 A-fragments
    FragBf qa[2];
    {
        const uint16_t* qp = qws + (((size_t)b2 * HEADS + h) * NTOK + q0 + l16) * DH;
#pragma unroll
        for (int kd = 0; kd < 2; ++kd)
#pragma unroll
            for (int i = 0; i < 8; ++i) {
                int kk = kd * 32 + ((i < 4) ? 0 : 16) + half * 8 + (i & 3) * 2;
                qa[kd].u[i] = *(const uint32_t*)(qp + kk);
            }
    }

    float m_run[8], l_run[8];
    v8f o_acc[4];
#pragma unroll
    for (int v = 0; v < 8; ++v) { m_run[v] = -1e30f; l_run[v] = 0.f; }
#pragma unroll
    for (int nt = 0; nt < 4; ++nt) o_acc[nt] = v8f_zero();

    // 32-key chunk schedule (uniform across block)
    int chB[16], chC[16], nch;
    if (qb == 0) {
        nch = 4;
        for (int i = 0; i < 4; ++i) { chB[i] = b2; chC[i] = i * 32; }
    } else {
        nch = 16;
        for (int i = 0; i < 4; ++i) { chB[i]     = b2 & 31;        chC[i]     = i * 32; }
        for (int i = 0; i < 4; ++i) { chB[4 + i] = (b2 & 31) + 32; chC[4 + i] = i * 32; }
        for (int i = 0; i < 8; ++i) { chB[8 + i] = b2;             chC[8 + i] = 128 + i * 32; }
    }

    // cooperative async staging of one chunk (K: 64x32, V: 32x64 bf16)
    auto issue = [&](int i, int buf) {
        const size_t kb = (size_t)chB[i] * HEADS + h;
        const int    c0 = chC[i];
        const uint16_t* kg = kws + (kb * DH + (t >> 2)) * NTOK + c0 + (t & 3) * 8;
        async_copy_b128(kg, &Kl[buf][(t >> 2) * 40 + (t & 3) * 8]);
        const uint16_t* vg = vws + (kb * NTOK + c0 + (t >> 3)) * DH + (t & 7) * 8;
        async_copy_b128(vg, &Vl[buf][(t >> 3) * 72 + (t & 7) * 8]);
    };

    uint16_t* pl = &Pl[wave * 16 * 32];

    issue(0, 0);
    for (int i = 0; i < nch; ++i) {
        wait_async0();          // chunk i landed (own loads)
        __syncthreads();        // ... for every wave
        if (i + 1 < nch) issue(i + 1, (i + 1) & 1);   // overlap with compute
        const int buf = i & 1;

        // ---- S = Q * K^T ----
        v8f s[2];
#pragma unroll
        for (int jt = 0; jt < 2; ++jt) {
            v8f sa = v8f_zero();
#pragma unroll
            for (int kd = 0; kd < 2; ++kd) {
                FragBf kb;
                const uint16_t* kp =
                    &Kl[buf][(kd * 32) * 40 + jt * 16 + l16];
#pragma unroll
                for (int j = 0; j < 8; ++j) {
                    int kk = ((j < 4) ? 0 : 16) + half * 8 + (j & 3) * 2;
                    kb.u[j] = (uint32_t)kp[kk * 40] | ((uint32_t)kp[(kk + 1) * 40] << 16);
                }
                sa = wmma_bf16(qa[kd], kb, sa);
            }
            s[jt] = sa;
        }
        // ---- online softmax (rows m = v + 8*half) ----
#pragma unroll
        for (int v = 0; v < 8; ++v) {
            float s0 = s[0][v] * scale;
            float s1 = s[1][v] * scale;
            float x = fmaxf(s0, s1);
#pragma unroll
            for (int msk = 1; msk < 16; msk <<= 1)
                x = fmaxf(x, __shfl_xor(x, msk, 32));
            float mnew = fmaxf(m_run[v], x);
            float ef = __expf(m_run[v] - mnew);
            float p0 = __expf(s0 - mnew);
            float p1 = __expf(s1 - mnew);
            float r = p0 + p1;
#pragma unroll
            for (int msk = 1; msk < 16; msk <<= 1)
                r += __shfl_xor(r, msk, 32);
            l_run[v] = l_run[v] * ef + r;
            m_run[v] = mnew;
#pragma unroll
            for (int nt = 0; nt < 4; ++nt) o_acc[nt][v] *= ef;
            int row = v + 8 * half;
            pl[row * 32 + l16]      = f2bfu(p0);
            pl[row * 32 + 16 + l16] = f2bfu(p1);
        }
        // ---- P @ V ----
        FragBf pa;
        {
            const uint16_t* pp = &pl[l16 * 32];
#pragma unroll
            for (int j = 0; j < 8; ++j) {
                int kk = ((j < 4) ? 0 : 16) + half * 8 + (j & 3) * 2;
                pa.u[j] = *(const uint32_t*)(pp + kk);
            }
        }
#pragma unroll
        for (int nt = 0; nt < 4; ++nt) {
            FragBf vb;
            const uint16_t* vp = &Vl[buf][nt * 16 + l16];
#pragma unroll
            for (int j = 0; j < 8; ++j) {
                int kk = ((j < 4) ? 0 : 16) + half * 8 + (j & 3) * 2;
                vb.u[j] = (uint32_t)vp[kk * 72] | ((uint32_t)vp[(kk + 1) * 72] << 16);
            }
            o_acc[nt] = wmma_bf16(pa, vb, o_acc[nt]);
        }
    }

    // ---- normalize, write attention output [row][C] bf16 ----
#pragma unroll
    for (int nt = 0; nt < 4; ++nt)
#pragma unroll
        for (int v = 0; v < 8; ++v) {
            int m = v + 8 * half;
            float val = o_acc[nt][v] / l_run[v];
            aws[((size_t)b2 * NTOK + q0 + m) * KC + h * DH + nt * 16 + l16] = f2bfu(val);
        }
}

// ---------------------------------------------------------------------------
// Kernel 3: out = attn @ proj_w^T + proj_b (fp32 out); async A staging.
// ---------------------------------------------------------------------------
__global__ __launch_bounds__(256) void proj_gemm_kernel(
    const uint16_t* __restrict__ aws, const float* __restrict__ proj_w,
    const float* __restrict__ proj_b, float* __restrict__ out)
{
    __shared__ __align__(16) uint16_t As[128 * 40];
    __shared__ __align__(16) uint16_t Bs[32 * 136];

    const int t    = threadIdx.x;
    const int lane = t & 31, wave = t >> 5;
    const int half = lane >> 4, l16 = lane & 15;
    const int wm   = wave >> 1, wn = wave & 1;
    const int bm   = blockIdx.x, bn = blockIdx.y;

    v8f acc[2][4];
#pragma unroll
    for (int mt = 0; mt < 2; ++mt)
#pragma unroll
        for (int nt = 0; nt < 4; ++nt) acc[mt][nt] = v8f_zero();

    const int ar   = t >> 3;
    const int ac   = (t & 7) * 4;
    const int arow = t >> 1;          // 0..127
    const int aoff = (t & 1) * 16;    // 0 or 16 halves

    for (int kt = 0; kt < 24; ++kt) {
        const int k0 = kt * 32;
        // ---- A: bf16, async copy to LDS (2 x b128 per thread) ----
        {
            const uint16_t* src = aws + (size_t)(bm * 128 + arow) * KC + k0 + aoff;
            async_copy_b128(src,     &As[arow * 40 + aoff]);
            async_copy_b128(src + 8, &As[arow * 40 + aoff + 8]);
        }
        // ---- B: proj_w fp32 -> bf16 transposed ----
#pragma unroll
        for (int nn = 0; nn < 4; ++nn) {
            int n  = ar + nn * 32;
            int gn = bn * 128 + n;
            const float* src = proj_w + (size_t)gn * KC + k0 + ac;
            float4 v = *(const float4*)src;
            Bs[(ac + 0) * 136 + n] = f2bfu(v.x);
            Bs[(ac + 1) * 136 + n] = f2bfu(v.y);
            Bs[(ac + 2) * 136 + n] = f2bfu(v.z);
            Bs[(ac + 3) * 136 + n] = f2bfu(v.w);
        }
        wait_async0();
        __syncthreads();

        FragBf a[2], b[4];
#pragma unroll
        for (int mt = 0; mt < 2; ++mt) {
            const uint16_t* ap = &As[(wm * 32 + mt * 16 + l16) * 40];
#pragma unroll
            for (int i = 0; i < 8; ++i) {
                int kk = ((i < 4) ? 0 : 16) + half * 8 + (i & 3) * 2;
                a[mt].u[i] = *(const uint32_t*)(ap + kk);
            }
        }
#pragma unroll
        for (int nt = 0; nt < 4; ++nt) {
            const uint16_t* bp = &Bs[wn * 64 + nt * 16 + l16];
#pragma unroll
            for (int i = 0; i < 8; ++i) {
                int kk = ((i < 4) ? 0 : 16) + half * 8 + (i & 3) * 2;
                b[nt].u[i] = (uint32_t)bp[kk * 136] | ((uint32_t)bp[(kk + 1) * 136] << 16);
            }
        }
#pragma unroll
        for (int mt = 0; mt < 2; ++mt)
#pragma unroll
            for (int nt = 0; nt < 4; ++nt)
                acc[mt][nt] = wmma_bf16(a[mt], b[nt], acc[mt][nt]);
        __syncthreads();
    }

#pragma unroll
    for (int mt = 0; mt < 2; ++mt)
#pragma unroll
        for (int nt = 0; nt < 4; ++nt) {
            int cg = bn * 128 + wn * 64 + nt * 16 + l16;
            float bias = proj_b[cg];
#pragma unroll
            for (int v = 0; v < 8; ++v) {
                int rg = bm * 128 + wm * 32 + mt * 16 + v + 8 * half;
                out[(size_t)rg * KC + cg] = acc[mt][nt][v] + bias;
            }
        }
}

// ---------------------------------------------------------------------------
extern "C" void kernel_launch(void* const* d_in, const int* in_sizes, int n_in,
                              void* d_out, int out_size, void* d_ws, size_t ws_size,
                              hipStream_t stream) {
    (void)in_sizes; (void)n_in; (void)out_size; (void)ws_size;
    const float* xv     = (const float*)d_in[0];
    const float* xi     = (const float*)d_in[1];
    const float* qkv_w  = (const float*)d_in[2];
    const float* proj_w = (const float*)d_in[3];
    const float* proj_b = (const float*)d_in[4];
    float* out = (float*)d_out;

    uint16_t* ws  = (uint16_t*)d_ws;
    uint16_t* qws = ws;
    uint16_t* kws = ws + SEG_ELEMS;
    uint16_t* vws = ws + 2 * SEG_ELEMS;
    uint16_t* aws = ws + 3 * SEG_ELEMS;

    qkv_gemm_kernel<<<dim3(192, 18), 256, 0, stream>>>(xv, xi, qkv_w, qws, kws, vws);
    attn_kernel<<<dim3(64 * HEADS * 3), 256, 0, stream>>>(qws, kws, vws, aws);
    proj_gemm_kernel<<<dim3(192, 6), 256, 0, stream>>>(aws, proj_w, proj_b, out);
}
